// AttnLayerV3_17403207483600
// MI455X (gfx1250) — matrix operand order repaired and checked by reading, not compiled, not run on hardware
//
#include <hip/hip_runtime.h>
#include <cstdint>
#include <cstddef>

typedef _Float16 v8h  __attribute__((ext_vector_type(8)));
typedef _Float16 v16h __attribute__((ext_vector_type(16)));
typedef float    v8f  __attribute__((ext_vector_type(8)));

#define DEV static __device__ __forceinline__

DEV v16h cat16(v8h lo, v8h hi) {
  return __builtin_shufflevector(lo, hi, 0,1,2,3,4,5,6,7,8,9,10,11,12,13,14,15);
}

// A-fragment (16x32 f16, M x K): lane L -> row (L&15);
// K = [h8, h8+8) U [16+h8, 16+h8+8), h8 = (L>>4)*8  -> two 16B contiguous loads
DEV v16h load_a(const _Float16* base, int stride, int lane) {
  const _Float16* p = base + (size_t)(lane & 15) * stride + ((lane >> 4) << 3);
  v8h lo = *(const v8h*)p;
  v8h hi = *(const v8h*)(p + 16);
  return cat16(lo, hi);
}

// B-fragment (32x16 f16, K x N) read from N-major (transposed) storage:
// lane L -> column (L&15) = row of Bt; K = [h8, h8+8) U [16+h8, 16+h8+8), h8 = (L>>4)*8
DEV v16h load_b(const _Float16* base, int stride, int lane) {
  const _Float16* p = base + (size_t)(lane & 15) * stride + ((lane >> 4) << 3);
  v8h lo = *(const v8h*)p;
  v8h hi = *(const v8h*)(p + 16);
  return cat16(lo, hi);
}

DEV v8f wmma16(v16h a, v16h b, v8f c) {
  return __builtin_amdgcn_wmma_f32_16x16x32_f16(false, a, false, b, (short)0, c, false, false);
}

// ---------------------------------------------------------------- small prep
__global__ void k_lambda(const float* __restrict__ lq1, const float* __restrict__ lk1,
                         const float* __restrict__ lq2, const float* __restrict__ lk2,
                         float* __restrict__ lam) {
  __shared__ float s1[128], s2[128];
  int i = threadIdx.x;
  s1[i] = lq1[i] * lk1[i];
  s2[i] = lq2[i] * lk2[i];
  __syncthreads();
  for (int off = 64; off > 0; off >>= 1) {
    if (i < off) { s1[i] += s1[i + off]; s2[i] += s2[i + off]; }
    __syncthreads();
  }
  if (i == 0) *lam = __expf(s1[0]) - __expf(s2[0]) + 0.2f; // LAMBDA_INIT = 0.2
}

__global__ void k_ropetab(float* __restrict__ cosT, float* __restrict__ sinT) {
  int t = blockIdx.x, i = threadIdx.x; // 64 threads
  float inv = __powf(10000.0f, -(float)i / 64.0f);
  float f = (float)t * inv;
  cosT[t * 64 + i] = cosf(f);
  sinT[t * 64 + i] = sinf(f);
}

__global__ void k_cvt16(const float* __restrict__ s, _Float16* __restrict__ d, int n) {
  int i = blockIdx.x * blockDim.x + threadIdx.x;
  if (i < n) d[i] = (_Float16)s[i];
}

// src (R x C) f32 row-major -> dst (C x R) f16 row-major
__global__ __launch_bounds__(256) void k_transpose16(const float* __restrict__ src,
                                                     _Float16* __restrict__ dst,
                                                     int R, int C) {
  __shared__ float tile[32][33];
  int tx = threadIdx.x, ty = threadIdx.y;           // 32 x 8
  int c0 = blockIdx.x * 32, r0 = blockIdx.y * 32;
#pragma unroll
  for (int i = 0; i < 32; i += 8)
    tile[ty + i][tx] = src[(size_t)(r0 + ty + i) * C + c0 + tx];
  __syncthreads();
#pragma unroll
  for (int i = 0; i < 32; i += 8)
    dst[(size_t)(c0 + ty + i) * R + r0 + tx] = (_Float16)tile[tx][ty + i];
}

// ---------------------------------------------------------------- WMMA GEMM
// C(MxN,f32) = A(MxK,f16 row-major) * Bt(NxK,f16 row-major)^T
// 2-stage software pipeline: tile k+1 loads issue before tile k's 16 WMMAs.
__global__ __launch_bounds__(128) void k_gemm(const _Float16* __restrict__ A,
                                              const _Float16* __restrict__ Bt,
                                              float* __restrict__ C,
                                              int M, int N, int K) {
  const int tid = threadIdx.x, lane = tid & 31, wid = tid >> 5;
  const int half = lane >> 4, l16 = lane & 15;
  const int m0 = blockIdx.y * 128 + (wid >> 1) * 64;
  const int n0 = blockIdx.x * 128 + (wid & 1) * 64;
  v8f acc[4][4] = {};
  v16h a[4], b[4];
#pragma unroll
  for (int i = 0; i < 4; ++i) a[i] = load_a(A + (size_t)(m0 + i * 16) * K, K, lane);
#pragma unroll
  for (int j = 0; j < 4; ++j) b[j] = load_b(Bt + (size_t)(n0 + j * 16) * K, K, lane);

  for (int kc = 0; kc + 32 < K; kc += 32) {
    const int kn = kc + 32;
    v16h an[4], bn[4];
#pragma unroll
    for (int i = 0; i < 4; ++i)
      an[i] = load_a(A + (size_t)(m0 + i * 16) * K + kn, K, lane);
#pragma unroll
    for (int j = 0; j < 4; ++j)
      bn[j] = load_b(Bt + (size_t)(n0 + j * 16) * K + kn, K, lane);
    __builtin_prefetch(A + (size_t)(m0 + l16) * K + kn + 32, 0, 1);
    __builtin_prefetch(Bt + (size_t)(n0 + l16) * K + kn + 32, 0, 1);
#pragma unroll
    for (int i = 0; i < 4; ++i)
#pragma unroll
      for (int j = 0; j < 4; ++j)
        acc[i][j] = wmma16(a[i], b[j], acc[i][j]);
#pragma unroll
    for (int i = 0; i < 4; ++i) { a[i] = an[i]; b[i] = bn[i]; }
  }
#pragma unroll
  for (int i = 0; i < 4; ++i)
#pragma unroll
    for (int j = 0; j < 4; ++j)
      acc[i][j] = wmma16(a[i], b[j], acc[i][j]);

#pragma unroll
  for (int i = 0; i < 4; ++i)
#pragma unroll
    for (int j = 0; j < 4; ++j)
#pragma unroll
      for (int g = 0; g < 8; ++g) {
        int row = m0 + i * 16 + half * 8 + g;
        int col = n0 + j * 16 + l16;
        C[(size_t)row * N + col] = acc[i][j][g];
      }
}

// ---------------------------------------------------------------- RoPE / split
__global__ void k_rope_q(const float* __restrict__ qraw, const float* __restrict__ cosT,
                         const float* __restrict__ sinT, _Float16* __restrict__ q1h,
                         _Float16* __restrict__ q2h) {
  int t = blockIdx.x, h = blockIdx.y, j = threadIdx.x; // 128 threads
  int set = j >> 6, i = j & 63;
  const float* base = qraw + (size_t)t * 4096 + h * 256 + set * 128;
  float x1 = base[2 * i], x2 = base[2 * i + 1];
  float c = cosT[t * 64 + i], s = sinT[t * 64 + i];
  _Float16* dst = (set ? q2h : q1h) + ((size_t)h * 2048 + t) * 128;
  dst[2 * i]     = (_Float16)(x1 * c - x2 * s);
  dst[2 * i + 1] = (_Float16)(x1 * s + x2 * c);
}

__global__ void k_rope_k(const float* __restrict__ kraw, const float* __restrict__ cosT,
                         const float* __restrict__ sinT, _Float16* __restrict__ k1h,
                         _Float16* __restrict__ k2h) {
  int t = blockIdx.x, kv = blockIdx.y, j = threadIdx.x; // 128 threads
  int set = j >> 6, i = j & 63;
  const float* base = kraw + (size_t)t * 1024 + kv * 256 + set * 128;
  float x1 = base[2 * i], x2 = base[2 * i + 1];
  float c = cosT[t * 64 + i], s = sinT[t * 64 + i];
  _Float16* dst = (set ? k2h : k1h) + ((size_t)kv * 2048 + t) * 128;
  dst[2 * i]     = (_Float16)(x1 * c - x2 * s);
  dst[2 * i + 1] = (_Float16)(x1 * s + x2 * c);
}

// v_raw (T x 1024 f32) -> vT[kv][dim(256)][T] f16
__global__ void k_vT(const float* __restrict__ vraw, _Float16* __restrict__ vT) {
  int rc = blockIdx.x; // kv*256 + c, 0..1023
  _Float16* dst = vT + (size_t)rc * 2048;
  for (int t = threadIdx.x; t < 2048; t += blockDim.x)
    dst[t] = (_Float16)vraw[(size_t)t * 1024 + rc];
}

// ---------------------------------------------------------------- attention
__global__ __launch_bounds__(128) void k_attn(const _Float16* __restrict__ q1h,
    const _Float16* __restrict__ q2h, const _Float16* __restrict__ k1h,
    const _Float16* __restrict__ k2h, const _Float16* __restrict__ vT,
    float* __restrict__ attn, float* __restrict__ partial,
    const float* __restrict__ lamp) {
  __shared__ __align__(16) _Float16 ldsP[4][2][16][40];
  __shared__ float redS[4], redQ[4];
  const int tid = threadIdx.x, lane = tid & 31, wid = tid >> 5;
  const int half = lane >> 4, l16 = lane & 15;
  const int qt = blockIdx.x, h = blockIdx.y;
  const int q0 = qt * 64 + wid * 16;
  const int kv = h >> 2;
  const float lam = *lamp;
  const float scale = 0.08838834764831845f; // 1/sqrt(128)

  const _Float16* q1p = q1h + ((size_t)h * 2048 + q0) * 128;
  const _Float16* q2p = q2h + ((size_t)h * 2048 + q0) * 128;
  const _Float16* k1p = k1h + (size_t)kv * 2048 * 128;
  const _Float16* k2p = k2h + (size_t)kv * 2048 * 128;
  const _Float16* vp  = vT  + (size_t)kv * 256 * 2048;

  v16h qa1[4], qa2[4];
#pragma unroll
  for (int d = 0; d < 4; ++d) {
    qa1[d] = load_a(q1p + d * 32, 128, lane);
    qa2[d] = load_a(q2p + d * 32, 128, lane);
  }

  v8f o1[16] = {}, o2[16] = {};
  float m1[8], l1[8], m2[8], l2[8];
#pragma unroll
  for (int g = 0; g < 8; ++g) { m1[g] = m2[g] = -__builtin_inff(); l1[g] = l2[g] = 0.f; }

  const int nkt = (q0 + 47) >> 5; // cover keys 0 .. q0+15
  for (int kt = 0; kt < nkt; ++kt) {
    const int kb = kt * 32;
    v8f s1[2] = {}, s2[2] = {};
#pragma unroll
    for (int sub = 0; sub < 2; ++sub) {
      const _Float16* kr1 = k1p + (size_t)(kb + sub * 16) * 128;
      const _Float16* kr2 = k2p + (size_t)(kb + sub * 16) * 128;
#pragma unroll
      for (int d = 0; d < 4; ++d) {
        s1[sub] = wmma16(qa1[d], load_b(kr1 + d * 32, 128, lane), s1[sub]);
        s2[sub] = wmma16(qa2[d], load_b(kr2 + d * 32, 128, lane), s2[sub]);
      }
    }
#pragma unroll
    for (int g = 0; g < 8; ++g) {
      const int tq = q0 + half * 8 + g;
      float a0 = s1[0][g] * scale; if (kb      + l16 > tq) a0 = -__builtin_inff();
      float a1 = s1[1][g] * scale; if (kb + 16 + l16 > tq) a1 = -__builtin_inff();
      float b0 = s2[0][g] * scale; if (kb      + l16 > tq) b0 = -__builtin_inff();
      float b1 = s2[1][g] * scale; if (kb + 16 + l16 > tq) b1 = -__builtin_inff();
      // ---- softmax stream 1
      float tm = fmaxf(a0, a1);
#pragma unroll
      for (int off = 1; off < 16; off <<= 1) tm = fmaxf(tm, __shfl_xor(tm, off, 32));
      float nm = fmaxf(m1[g], tm);
      float r  = __expf(m1[g] - nm);
      m1[g] = nm;
      float e0 = __expf(a0 - nm), e1 = __expf(a1 - nm);
      float rs = e0 + e1;
#pragma unroll
      for (int off = 1; off < 16; off <<= 1) rs += __shfl_xor(rs, off, 32);
      l1[g] = l1[g] * r + rs;
#pragma unroll
      for (int nt = 0; nt < 16; ++nt) o1[nt][g] = o1[nt][g] * r;
      ldsP[wid][0][half * 8 + g][l16]      = (_Float16)e0;
      ldsP[wid][0][half * 8 + g][16 + l16] = (_Float16)e1;
      // ---- softmax stream 2
      tm = fmaxf(b0, b1);
#pragma unroll
      for (int off = 1; off < 16; off <<= 1) tm = fmaxf(tm, __shfl_xor(tm, off, 32));
      nm = fmaxf(m2[g], tm);
      r  = __expf(m2[g] - nm);
      m2[g] = nm;
      e0 = __expf(b0 - nm); e1 = __expf(b1 - nm);
      rs = e0 + e1;
#pragma unroll
      for (int off = 1; off < 16; off <<= 1) rs += __shfl_xor(rs, off, 32);
      l2[g] = l2[g] * r + rs;
#pragma unroll
      for (int nt = 0; nt < 16; ++nt) o2[nt][g] = o2[nt][g] * r;
      ldsP[wid][1][half * 8 + g][l16]      = (_Float16)e0;
      ldsP[wid][1][half * 8 + g][16 + l16] = (_Float16)e1;
    }
    // C-layout -> A-layout relayout through wave-private LDS (in-order DS, same wave)
    v16h pa1 = load_a(&ldsP[wid][0][0][0], 40, lane);
    v16h pa2 = load_a(&ldsP[wid][1][0][0], 40, lane);
#pragma unroll
    for (int nt = 0; nt < 16; ++nt) {
      v16h vb = load_b(vp + (size_t)(nt * 16) * 2048 + kb, 2048, lane);
      o1[nt] = wmma16(pa1, vb, o1[nt]);
      o2[nt] = wmma16(pa2, vb, o2[nt]);
    }
  }

  float inv1[8], inv2[8];
#pragma unroll
  for (int g = 0; g < 8; ++g) { inv1[g] = 1.f / l1[g]; inv2[g] = 1.f / l2[g]; }
  float S = 0.f, Q = 0.f;
#pragma unroll
  for (int nt = 0; nt < 16; ++nt)
#pragma unroll
    for (int g = 0; g < 8; ++g) {
      float val = o1[nt][g] * inv1[g] - lam * (o2[nt][g] * inv2[g]);
      int t = q0 + half * 8 + g;
      attn[(size_t)t * 4096 + h * 256 + nt * 16 + l16] = val;
      S += val; Q += val * val;
    }
#pragma unroll
  for (int off = 16; off; off >>= 1) { S += __shfl_xor(S, off, 32); Q += __shfl_xor(Q, off, 32); }
  if (lane == 0) { redS[wid] = S; redQ[wid] = Q; }
  __syncthreads();
  if (tid == 0) {
    partial[(h * 32 + qt) * 2]     = redS[0] + redS[1] + redS[2] + redS[3];
    partial[(h * 32 + qt) * 2 + 1] = redQ[0] + redQ[1] + redQ[2] + redQ[3];
  }
}

// ---------------------------------------------------------------- norm
__global__ void k_stats(const float* __restrict__ partial, float* __restrict__ stats) {
  int h = threadIdx.x >> 5, i = threadIdx.x & 31; // 512 threads
  float S = partial[(h * 32 + i) * 2];
  float Q = partial[(h * 32 + i) * 2 + 1];
#pragma unroll
  for (int off = 16; off; off >>= 1) { S += __shfl_xor(S, off, 32); Q += __shfl_xor(Q, off, 32); }
  if (i == 0) {
    const float n = 2048.0f * 256.0f;
    float mean = S / n;
    float var  = Q / n - mean * mean;
    stats[h * 2]     = mean;
    stats[h * 2 + 1] = rsqrtf(var + 1e-5f);
  }
}

__global__ void k_norm(const float* __restrict__ attn, const float* __restrict__ stats,
                       const float* __restrict__ gw, const float* __restrict__ gb,
                       _Float16* __restrict__ y) {
  int i = blockIdx.x * blockDim.x + threadIdx.x; // exactly 2048*4096 threads
  int c = i & 4095, h = c >> 8;
  float mean = stats[h * 2], rstd = stats[h * 2 + 1];
  y[i] = (_Float16)(((attn[i] - mean) * rstd * gw[c] + gb[c]) * 0.8f); // *(1-LAMBDA_INIT)
}

// ---------------------------------------------------------------- launch
extern "C" void kernel_launch(void* const* d_in, const int* in_sizes, int n_in,
                              void* d_out, int out_size, void* d_ws, size_t ws_size,
                              hipStream_t stream) {
  (void)in_sizes; (void)n_in; (void)out_size; (void)ws_size;
  const float* x   = (const float*)d_in[0];
  const float* Wq  = (const float*)d_in[1];
  const float* Wk  = (const float*)d_in[2];
  const float* Wv  = (const float*)d_in[3];
  const float* Wo  = (const float*)d_in[4];
  const float* lq1 = (const float*)d_in[5];
  const float* lk1 = (const float*)d_in[6];
  const float* lq2 = (const float*)d_in[7];
  const float* lk2 = (const float*)d_in[8];
  const float* gw  = (const float*)d_in[9];
  const float* gb  = (const float*)d_in[10];

  char* w = (char*)d_ws;
  size_t off = 0;
  auto alloc = [&](size_t n) -> char* {
    off = (off + 255) & ~(size_t)255;
    char* p = w + off;
    off += n;
    return p;
  };
  float*     lam   = (float*)alloc(4);
  float*     stats = (float*)alloc(16 * 2 * 4);
  float*     part  = (float*)alloc(16 * 32 * 2 * 4);
  float*     cosT  = (float*)alloc(2048 * 64 * 4);
  float*     sinT  = (float*)alloc(2048 * 64 * 4);
  _Float16*  x16   = (_Float16*)alloc((size_t)2048 * 2048 * 2);
  _Float16*  WqT   = (_Float16*)alloc((size_t)4096 * 2048 * 2);
  _Float16*  WkT   = (_Float16*)alloc((size_t)1024 * 2048 * 2);
  _Float16*  WvT   = (_Float16*)alloc((size_t)1024 * 2048 * 2);
  _Float16*  WoT   = (_Float16*)alloc((size_t)2048 * 4096 * 2);
  float*     qraw  = (float*)alloc((size_t)2048 * 4096 * 4);
  float*     kraw  = (float*)alloc((size_t)2048 * 1024 * 4);
  float*     vraw  = (float*)alloc((size_t)2048 * 1024 * 4);
  _Float16*  q1h   = (_Float16*)alloc((size_t)16 * 2048 * 128 * 2);
  _Float16*  q2h   = (_Float16*)alloc((size_t)16 * 2048 * 128 * 2);
  _Float16*  k1h   = (_Float16*)alloc((size_t)4 * 2048 * 128 * 2);
  _Float16*  k2h   = (_Float16*)alloc((size_t)4 * 2048 * 128 * 2);
  _Float16*  vTp   = (_Float16*)alloc((size_t)4 * 256 * 2048 * 2);
  // aliases (life-times disjoint): attn reuses qraw; y16 reuses kraw+vraw (contiguous 16MB)
  float*    attn = qraw;
  _Float16* y16  = (_Float16*)kraw;

  k_lambda<<<1, 128, 0, stream>>>(lq1, lk1, lq2, lk2, lam);
  k_ropetab<<<2048, 64, 0, stream>>>(cosT, sinT);
  k_cvt16<<<(2048 * 2048) / 256, 256, 0, stream>>>(x, x16, 2048 * 2048);
  k_transpose16<<<dim3(4096 / 32, 2048 / 32), dim3(32, 8), 0, stream>>>(Wq, WqT, 2048, 4096);
  k_transpose16<<<dim3(1024 / 32, 2048 / 32), dim3(32, 8), 0, stream>>>(Wk, WkT, 2048, 1024);
  k_transpose16<<<dim3(1024 / 32, 2048 / 32), dim3(32, 8), 0, stream>>>(Wv, WvT, 2048, 1024);
  k_transpose16<<<dim3(2048 / 32, 4096 / 32), dim3(32, 8), 0, stream>>>(Wo, WoT, 4096, 2048);

  k_gemm<<<dim3(4096 / 128, 2048 / 128), 128, 0, stream>>>(x16, WqT, qraw, 2048, 4096, 2048);
  k_gemm<<<dim3(1024 / 128, 2048 / 128), 128, 0, stream>>>(x16, WkT, kraw, 2048, 1024, 2048);
  k_gemm<<<dim3(1024 / 128, 2048 / 128), 128, 0, stream>>>(x16, WvT, vraw, 2048, 1024, 2048);

  k_rope_q<<<dim3(2048, 16), 128, 0, stream>>>(qraw, cosT, sinT, q1h, q2h);
  k_rope_k<<<dim3(2048, 4), 128, 0, stream>>>(kraw, cosT, sinT, k1h, k2h);
  k_vT<<<1024, 256, 0, stream>>>(vraw, vTp);

  k_attn<<<dim3(32, 16), 128, 0, stream>>>(q1h, q2h, k1h, k2h, vTp, attn, part, lam);
  k_stats<<<1, 512, 0, stream>>>(part, stats);
  k_norm<<<(2048 * 4096) / 1024, 1024, 0, stream>>>(attn, stats, gw, gb, y16);

  k_gemm<<<dim3(2048 / 128, 2048 / 128), 128, 0, stream>>>(y16, WoT, (float*)d_out,
                                                           2048, 2048, 4096);
}
